// VOSPRaftUpsampler_38706245272274
// MI455X (gfx1250) — matrix operand order repaired and checked by reading, not compiled, not run on hardware
//
#include <hip/hip_runtime.h>

typedef __attribute__((ext_vector_type(16))) _Float16 v16h;
typedef __attribute__((ext_vector_type(8)))  _Float16 v8h;
typedef __attribute__((ext_vector_type(8)))  float    v8f;

#define WAVES_PER_BLOCK 8   // 256 threads = 8 wave32

// ---------------- prep: NCHW f32 -> NHWC f16 ----------------
__global__ __launch_bounds__(256) void prep_x_nhwc_f16(
    const float* __restrict__ x, _Float16* __restrict__ xh, int N, int C, int H, int W) {
  size_t idx = (size_t)blockIdx.x * blockDim.x + threadIdx.x;
  size_t total = (size_t)N * C * H * W;
  if (idx >= total) return;
  int c = (int)(idx % C);
  size_t pix = idx / C;
  int w = (int)(pix % W);
  size_t t = pix / W;
  int h = (int)(t % H);
  int n = (int)(t / H);
  xh[idx] = (_Float16)x[(((size_t)n * C + c) * H + h) * W + w];
}

// ---- prep: OIHW f32 weights -> [CoutPad][Kpad] f16, K = tap*Cin + ci ----
__global__ __launch_bounds__(256) void prep_w_f16(
    const float* __restrict__ src, _Float16* __restrict__ dst,
    int Cout, int Cin, int KH, int KW, int CoutPad, int Kpad) {
  int idx = blockIdx.x * blockDim.x + threadIdx.x;
  if (idx >= CoutPad * Kpad) return;
  int kk = idx % Kpad;
  int co = idx / Kpad;
  int Ktot = Cin * KH * KW;
  float v = 0.f;
  if (co < Cout && kk < Ktot) {
    int tap = kk / Cin, ci = kk % Cin;
    int kh = tap / KW, kw = tap % KW;
    v = src[(((size_t)co * Cin + ci) * KH + kh) * KW + kw];
  }
  dst[idx] = (_Float16)v;
}

// ------------- LDS-staged implicit-GEMM conv via v_wmma_f32_16x16x32_f16 -------------
// Block = 8 waves = one 128-pixel row segment; activation tile (KH rows x 128+KW-1 cols
// x CIN ch, zero-padded halo) staged once in LDS; each wave computes a 16co x 16px tile.
// All K-decomposition constants are template params: divisions fold to shifts, the
// K-chunk loop fully unrolls, and the inner loop has zero bounds checks.
template <int CIN, int KH_, int KW_, int PAD, int KCHUNKS, int KPAD, int KTOT,
          bool RELU, bool OUTNCHW>
__global__ __launch_bounds__(256) void conv_wmma_t(
    const _Float16* __restrict__ act, const _Float16* __restrict__ wk,
    const float* __restrict__ bias, void* __restrict__ outp,
    int N, int H, int W, int Cout, int Cstore) {
  constexpr int BPX = 128;                 // pixels per block (row segment)
  constexpr int WT  = BPX + KW_ - 1;       // staged tile width incl. halo
  constexpr int FILL8 = KH_ * WT * CIN / 8;
  __shared__ __align__(16) _Float16 smem[KH_ * WT * CIN];

  int tid = threadIdx.x;
  int segsPerRow = W >> 7;
  int h   = blockIdx.x / segsPerRow;
  int seg = blockIdx.x - h * segsPerRow;
  int nCoG = (Cout + 15) >> 4;
  int n   = blockIdx.y / nCoG;
  int cog = blockIdx.y - n * nCoG;

  // ---- cooperative stage: 16B (8-half) vector moves, padding resolved here ----
  for (int i = tid; i < FILL8; i += 256) {
    int e    = i * 8;
    int ci   = e % CIN;                    // CIN is a power of two -> masks/shifts
    int rest = e / CIN;
    int col  = rest % WT;
    int row  = rest / WT;
    int y = h + row - PAD;
    int x = seg * BPX + col - PAD;
    v8h v;
#pragma unroll
    for (int j = 0; j < 8; ++j) v[j] = (_Float16)0.f;
    if (y >= 0 && y < H && x >= 0 && x < W)
      v = *(const v8h*)(act + (((size_t)n * H + y) * W + x) * CIN + ci);
    *(v8h*)(smem + e) = v;
  }
  __syncthreads();

  int wave = tid >> 5, lane = tid & 31;
  int ln  = lane & 15;
  int hi8 = (lane & 16) >> 1;  // A-fragment K base / D channel offset (0 or 8)
  int hiB = lane & 16;         // unused; B-fragment K base is hi8, as for A
  const _Float16* wrow = wk + (size_t)(cog * 16 + ln) * KPAD;

  v8f acc = {};
#pragma unroll
  for (int c = 0; c < KCHUNKS; ++c) {
    // A fragment: weights, two contiguous 16B runs from global (L2-resident)
    v16h av;
    {
      v8h a0 = *(const v8h*)(wrow + c * 32 + hi8);
      v8h a1 = *(const v8h*)(wrow + c * 32 + hi8 + 16);
#pragma unroll
      for (int i = 0; i < 8; ++i) { av[i] = a0[i]; av[i + 8] = a1[i]; }
    }
    // B fragment: two 16B ds_load_b128 runs from the staged LDS tile
    v16h bvv;
#pragma unroll
    for (int r = 0; r < 2; ++r) {
      int kb = c * 32 + hi8 + r * 16;
      v8h bv;
#pragma unroll
      for (int j = 0; j < 8; ++j) bv[j] = (_Float16)0.f;
      if (kb < KTOT) {                     // compile-time except boundary chunks
        int tap = kb / CIN;                // shift (CIN = pow2)
        int ci  = kb & (CIN - 1);
        int kh  = tap / KW_;               // constant-divisor mul/shift
        int kw  = tap - kh * KW_;
        int col = wave * 16 + ln + kw;     // halo origin already at -PAD
        bv = *(const v8h*)(smem + ((kh * WT + col) * CIN + ci));
      }
#pragma unroll
      for (int j = 0; j < 8; ++j) bvv[r * 8 + j] = bv[j];
    }
    acc = __builtin_amdgcn_wmma_f32_16x16x32_f16(false, av, false, bvv,
                                                 (short)0, acc, false, false);
  }

  // ---- epilogue: bias + ReLU + store ----
  float vals[8];
#pragma unroll
  for (int r = 0; r < 8; ++r) {
    int co = cog * 16 + hi8 + r;
    float v = acc[r] + (co < Cout ? bias[co] : 0.f);
    if (RELU) v = fmaxf(v, 0.f);
    vals[r] = v;
  }
  int px = seg * BPX + wave * 16 + ln;
  if (OUTNCHW) {                            // final output, f32 NCHW
    float* o = (float*)outp;
#pragma unroll
    for (int r = 0; r < 8; ++r) {
      int co = cog * 16 + hi8 + r;
      if (co < Cout)
        o[(((size_t)n * Cout + co) * H + h) * W + px] = vals[r];
    }
  } else {                                  // intermediate, f16 NHWC: one 16B store
    int cbase = cog * 16 + hi8;
    if (cbase < Cstore) {
      v8h hv;
#pragma unroll
      for (int r = 0; r < 8; ++r) hv[r] = (_Float16)vals[r];
      _Float16* o = (_Float16*)outp;
      *(v8h*)(o + (((size_t)n * H + h) * W + px) * Cstore + cbase) = hv;
    }
  }
}

// ------- guidance 1x1 conv (32 -> 144) with register-resident 9-way softmax -------
// D-tile for channel group k covers all 16 (p,q) at one window position k,
// so softmax over k is element-wise across the 9 accumulators.
__global__ __launch_bounds__(256) void gw_softmax_wmma(
    const _Float16* __restrict__ g, const _Float16* __restrict__ wk,
    const float* __restrict__ bias, _Float16* __restrict__ wts, int N, int H, int W) {
  const int Cin = 32, Kpad = 32;
  int lane = threadIdx.x & 31;
  int wave = threadIdx.x >> 5;
  int stripsPerRow = W >> 4;
  int tile = blockIdx.x * WAVES_PER_BLOCK + wave;
  int h  = tile / stripsPerRow;
  int w0 = (tile % stripsPerRow) << 4;
  int n  = blockIdx.y;
  int ln  = lane & 15;
  int hi8 = (lane & 16) >> 1;
  int hiB = lane & 16;

  v16h bvv;
  {
    const _Float16* brow = g + (((size_t)n * H + h) * W + (w0 + ln)) * Cin;
    v8h b0 = *(const v8h*)(brow + hi8);
    v8h b1 = *(const v8h*)(brow + hi8 + 16);
#pragma unroll
    for (int i = 0; i < 8; ++i) { bvv[i] = b0[i]; bvv[i + 8] = b1[i]; }
  }

  v8f acc[9];
#pragma unroll
  for (int k = 0; k < 9; ++k) {
    v16h av;
    const _Float16* wrow = wk + (size_t)(k * 16 + ln) * Kpad;
    v8h a0 = *(const v8h*)(wrow + hi8);
    v8h a1 = *(const v8h*)(wrow + hi8 + 16);
#pragma unroll
    for (int i = 0; i < 8; ++i) { av[i] = a0[i]; av[i + 8] = a1[i]; }
    v8f z = {};
    acc[k] = __builtin_amdgcn_wmma_f32_16x16x32_f16(false, av, false, bvv,
                                                    (short)0, z, false, false);
  }

#pragma unroll
  for (int r = 0; r < 8; ++r) {
    float e[9];
    float m = -3.4e38f;
#pragma unroll
    for (int k = 0; k < 9; ++k) {
      float v = acc[k][r] + bias[k * 16 + hi8 + r];
      e[k] = v;
      m = fmaxf(m, v);
    }
    float s = 0.f;
#pragma unroll
    for (int k = 0; k < 9; ++k) { e[k] = __expf(e[k] - m); s += e[k]; }
    float inv = 1.f / s;
#pragma unroll
    for (int k = 0; k < 9; ++k) acc[k][r] = e[k] * inv;
  }

  _Float16* orow = wts + (((size_t)n * H + h) * W + (w0 + ln)) * 144;
#pragma unroll
  for (int k = 0; k < 9; ++k) {
    v8h hv;
#pragma unroll
    for (int r = 0; r < 8; ++r) hv[r] = (_Float16)acc[k][r];
    *(v8h*)(orow + k * 16 + hi8) = hv;
  }
}

// ------- convex combination + pixel shuffle (K=9 -> VALU, memory-bound) -------
__global__ __launch_bounds__(256) void combine_pixelshuffle(
    const _Float16* __restrict__ xp, const _Float16* __restrict__ wts,
    _Float16* __restrict__ up, int N, int H, int W) {
  size_t idx = (size_t)blockIdx.x * blockDim.x + threadIdx.x;
  size_t total = (size_t)N * H * W;
  if (idx >= total) return;
  int w = (int)(idx % W);
  size_t t = idx / W;
  int h = (int)(t % H);
  int n = (int)(t / H);

  v8h pl[9], ph[9];
#pragma unroll
  for (int tap = 0; tap < 9; ++tap) {
    int y = h + tap / 3 - 1;
    int x = w + tap % 3 - 1;
    v8h z;
#pragma unroll
    for (int i = 0; i < 8; ++i) z[i] = (_Float16)0.f;
    pl[tap] = z; ph[tap] = z;
    if (y >= 0 && y < H && x >= 0 && x < W) {
      const _Float16* src = xp + (((size_t)n * H + y) * W + x) * 16;
      pl[tap] = *(const v8h*)(src);
      ph[tap] = *(const v8h*)(src + 8);
    }
  }

  const _Float16* wrow = wts + idx * 144;
  size_t H4 = (size_t)H * 4, W4 = (size_t)W * 4;
#pragma unroll
  for (int pq = 0; pq < 16; ++pq) {
    float a[16];
#pragma unroll
    for (int c = 0; c < 16; ++c) a[c] = 0.f;
#pragma unroll
    for (int k = 0; k < 9; ++k) {
      float wv = (float)wrow[k * 16 + pq];
#pragma unroll
      for (int c = 0; c < 16; ++c) {
        float pv = (float)((c < 8) ? pl[k][c] : ph[k][c - 8]);
        a[c] += wv * pv;
      }
    }
    int pp = pq >> 2, q = pq & 3;
    v8h o0, o1;
#pragma unroll
    for (int c = 0; c < 8; ++c) { o0[c] = (_Float16)a[c]; o1[c] = (_Float16)a[c + 8]; }
    _Float16* dst = up + (((size_t)n * H4 + (size_t)(4 * h + pp)) * W4 + (size_t)(4 * w + q)) * 16;
    *(v8h*)(dst)     = o0;
    *(v8h*)(dst + 8) = o1;
  }
}

// ------------------------------- host launcher -------------------------------
extern "C" void kernel_launch(void* const* d_in, const int* in_sizes, int n_in,
                              void* d_out, int out_size, void* d_ws, size_t ws_size,
                              hipStream_t stream) {
  (void)in_sizes; (void)n_in; (void)out_size; (void)ws_size;

  const float* x    = (const float*)d_in[0];
  const float* p1w1 = (const float*)d_in[1];
  const float* p1b1 = (const float*)d_in[2];
  const float* p1w2 = (const float*)d_in[3];
  const float* p1b2 = (const float*)d_in[4];
  const float* gww1 = (const float*)d_in[5];
  const float* gwb1 = (const float*)d_in[6];
  const float* gww2 = (const float*)d_in[7];
  const float* gwb2 = (const float*)d_in[8];
  const float* p2w1 = (const float*)d_in[9];
  const float* p2b1 = (const float*)d_in[10];
  const float* p2w2 = (const float*)d_in[11];
  const float* p2b2 = (const float*)d_in[12];

  const int N = 8, C = 64, H = 128, W = 128, H4 = 512, W4 = 512;

  char* ws = (char*)d_ws;
  size_t off = 0;
  auto alloc = [&](size_t bytes) -> char* {
    char* p = ws + off;
    off += (bytes + 255) & ~(size_t)255;
    return p;
  };
  _Float16* xh  = (_Float16*)alloc((size_t)N * H * W * 64 * 2);
  _Float16* t1  = (_Float16*)alloc((size_t)N * H * W * 32 * 2);
  _Float16* gg  = (_Float16*)alloc((size_t)N * H * W * 32 * 2);
  _Float16* xp  = (_Float16*)alloc((size_t)N * H * W * 16 * 2);
  _Float16* wts = (_Float16*)alloc((size_t)N * H * W * 144 * 2);
  _Float16* up  = (_Float16*)alloc((size_t)N * H4 * W4 * 16 * 2);
  _Float16* t2  = (_Float16*)alloc((size_t)N * H4 * W4 * 8 * 2);
  _Float16* wk1 = (_Float16*)alloc((size_t)32  * 576 * 2);
  _Float16* wk2 = (_Float16*)alloc((size_t)16  * 288 * 2);
  _Float16* wk3 = (_Float16*)alloc((size_t)32  * 576 * 2);
  _Float16* wk4 = (_Float16*)alloc((size_t)144 * 32  * 2);
  _Float16* wk5 = (_Float16*)alloc((size_t)16  * 416 * 2);
  _Float16* wk6 = (_Float16*)alloc((size_t)64  * 96  * 2);

  // --- prep ---
  {
    size_t tot = (size_t)N * C * H * W;
    prep_x_nhwc_f16<<<dim3((unsigned)((tot + 255) / 256)), dim3(256), 0, stream>>>(x, xh, N, C, H, W);
  }
  prep_w_f16<<<dim3((32 * 576 + 255) / 256), dim3(256), 0, stream>>>(p1w1, wk1, 32, 64, 3, 3, 32, 576);
  prep_w_f16<<<dim3((16 * 288 + 255) / 256), dim3(256), 0, stream>>>(p1w2, wk2, 16, 32, 3, 3, 16, 288);
  prep_w_f16<<<dim3((32 * 576 + 255) / 256), dim3(256), 0, stream>>>(gww1, wk3, 32, 64, 3, 3, 32, 576);
  prep_w_f16<<<dim3((144 * 32 + 255) / 256), dim3(256), 0, stream>>>(gww2, wk4, 144, 32, 1, 1, 144, 32);
  prep_w_f16<<<dim3((16 * 416 + 255) / 256), dim3(256), 0, stream>>>(p2w1, wk5, 8, 16, 5, 5, 16, 416);
  prep_w_f16<<<dim3((64 * 96 + 255) / 256), dim3(256), 0, stream>>>(p2w2, wk6, 64, 8, 3, 3, 64, 96);

  // grid: x = row-segments (128 px each), y = N * ceil(Cout/16)
  // project1 conv1: 64 -> 32, k3, ReLU
  conv_wmma_t<64, 3, 3, 1, 18, 576, 576, true, false>
      <<<dim3(H * (W >> 7), N * 2), dim3(256), 0, stream>>>(
      xh, wk1, p1b1, t1, N, H, W, 32, 32);
  // project1 conv2: 32 -> 16, k3, ReLU
  conv_wmma_t<32, 3, 3, 1, 9, 288, 288, true, false>
      <<<dim3(H * (W >> 7), N * 1), dim3(256), 0, stream>>>(
      t1, wk2, p1b2, xp, N, H, W, 16, 16);
  // guidance conv1: 64 -> 32, k3, ReLU
  conv_wmma_t<64, 3, 3, 1, 18, 576, 576, true, false>
      <<<dim3(H * (W >> 7), N * 2), dim3(256), 0, stream>>>(
      xh, wk3, gwb1, gg, N, H, W, 32, 32);
  // guidance conv2 (1x1, 32 -> 144) + fused softmax over 9-window
  gw_softmax_wmma<<<dim3(H * (W >> 4) / WAVES_PER_BLOCK, N), dim3(256), 0, stream>>>(
      gg, wk4, gwb2, wts, N, H, W);
  // convex combination + pixel shuffle
  {
    size_t tot = (size_t)N * H * W;
    combine_pixelshuffle<<<dim3((unsigned)((tot + 255) / 256)), dim3(256), 0, stream>>>(
        xp, wts, up, N, H, W);
  }
  // project2 conv1: 16 -> 8, k5 pad2, ReLU (Cout padded to 16-row WMMA tile, stores 8ch)
  conv_wmma_t<16, 5, 5, 2, 13, 416, 400, true, false>
      <<<dim3(H4 * (W4 >> 7), N * 1), dim3(256), 0, stream>>>(
      up, wk5, p2b1, t2, N, H4, W4, 8, 8);
  // project2 conv2: 8 -> 64, k3, no ReLU, write f32 NCHW into d_out
  conv_wmma_t<8, 3, 3, 1, 3, 96, 72, false, true>
      <<<dim3(H4 * (W4 >> 7), N * 4), dim3(256), 0, stream>>>(
      t2, wk6, p2b2, d_out, N, H4, W4, 64, 64);
}
